// CustomGRUCell_62027917689188
// MI455X (gfx1250) — compile-verified, matrix operand order checked by reading, not run on hardware
//
#include <hip/hip_runtime.h>
#include <hip/hip_bf16.h>
#include <math.h>

// ---------------------------------------------------------------------------
// GRU cell on MI455X: 3x bf16-WMMA GEMMs fed by the Tensor Data Mover,
// fp32 accumulate, fused sigmoid/tanh epilogue.
//   xproj = x @ W_ih^T + b_ih   (4096 x 3072)
//   rzh   = h @ W_rzh^T         (4096 x 2048)
//   nh    = h @ W_nh^T + b_nh   (4096 x 1024)
// ---------------------------------------------------------------------------

typedef __attribute__((ext_vector_type(16))) __bf16 v16bf;
typedef __attribute__((ext_vector_type(8)))  __bf16 v8bf;
typedef __attribute__((ext_vector_type(8)))  float  v8f;
typedef __attribute__((ext_vector_type(4)))  unsigned v4u;
typedef __attribute__((ext_vector_type(8)))  unsigned v8u;

#define B_DIM  4096
#define IN_DIM 1024
#define H_DIM  1024

#define BM 128
#define BN 128
#define BK 32
#define LDSS 40   // LDS row stride in bf16 elems: 32 data + 8 pad (80B, 16B aligned)

// ----------------------------- fp32 -> bf16 --------------------------------
__device__ __forceinline__ unsigned short f32_to_bf16_rne(float f) {
  union { float f; unsigned u; } c; c.f = f;
  unsigned u = c.u;
  return (unsigned short)((u + 0x7FFFu + ((u >> 16) & 1u)) >> 16);
}

__global__ void convert_f32_bf16_kernel(const float* __restrict__ in,
                                        unsigned short* __restrict__ out,
                                        int n4) {
  int idx = blockIdx.x * blockDim.x + threadIdx.x;
  if (idx >= n4) return;
  float4 v = ((const float4*)in)[idx];
  ushort4 o;
  o.x = f32_to_bf16_rne(v.x);
  o.y = f32_to_bf16_rne(v.y);
  o.z = f32_to_bf16_rne(v.z);
  o.w = f32_to_bf16_rne(v.w);
  ((ushort4*)out)[idx] = o;
}

// --------------------- TDM: DMA one 16-row x 32-col bf16 tile slice --------
// Descriptor per CDNA5 ISA:
//   group0: count=1 | lds_addr | global_addr(57b) | type=2
//   group1: data_size=2B, pad_enable, pad_interval=16 DWORDs (64B = one 32-elem
//           row), pad_amount=4 DWORDs (16B) -> LDS row stride 40 elems = LDSS.
//   tile_dim0=32 (K step), tile_dim1=16 (rows), tensor_dim0_stride=K.
__device__ __forceinline__ void tdm_load_tile_bf16(
    const unsigned short* gtile,   // global address of tile start
    unsigned lds_byte_addr,        // LDS destination byte address
    unsigned tensor_rows,          // tensor_dim1 (OOB bound; tiles are interior)
    unsigned k_elems) {            // tensor_dim0 == row stride in elements
  unsigned long long ga = (unsigned long long)gtile;
  v4u g0;
  g0.x = 1u;                                                  // count=1 (valid)
  g0.y = lds_byte_addr;                                       // lds_addr
  g0.z = (unsigned)ga;                                        // global_addr lo
  g0.w = ((unsigned)(ga >> 32) & 0x01FFFFFFu) | (2u << 30);   // addr hi | type=2
  v8u g1;
  g1.s0 = (1u << 16)      // data_size = 1 -> 2 bytes
        | (1u << 20)      // pad_enable
        | (3u << 22)      // pad_interval: 16 DWORDs
        | (3u << 25);     // pad_amount:   4 DWORDs
  g1.s1 = (k_elems & 0xFFFFu) << 16;                              // dim0 lo16
  g1.s2 = ((k_elems >> 16) & 0xFFFFu) | ((tensor_rows & 0xFFFFu) << 16);
  g1.s3 = ((tensor_rows >> 16) & 0xFFFFu) | (32u << 16);          // tile_dim0=32
  g1.s4 = 16u;                                                    // tile_dim1=16
  g1.s5 = k_elems;                                                // dim0_stride lo
  g1.s6 = 0u;
  g1.s7 = 0u;
  asm volatile("tensor_load_to_lds %0, %1" :: "s"(g0), "s"(g1) : "memory");
}

// --------------------- bf16 WMMA GEMM: C = A * W^T + bias ------------------
// A: [M,K] bf16 row-major.  W: [N,K] bf16 row-major (dot rows with rows).
// Block 128x128, 8 waves, wave tile 32x64 (2 M-frags x 4 N-frags), BK=32.
// Tile feed: per-wave TDM slices, TENSORcnt + one barrier per K-step.
__global__ __launch_bounds__(256)
void gemm_bf16_wmma_kernel(const unsigned short* __restrict__ A,
                           const unsigned short* __restrict__ W,
                           const float* __restrict__ bias,   // may be null
                           float* __restrict__ C,
                           int M, int N, int K) {
  __shared__ __align__(16) unsigned short smA[2][BM * LDSS];
  __shared__ __align__(16) unsigned short smB[2][BN * LDSS];

  const int tid  = threadIdx.x;
  const int lane = tid & 31;
  const int l16  = lane & 15;
  const int hk   = lane >> 4;                              // K-interleave half
  const int wv   = __builtin_amdgcn_readfirstlane(tid) >> 5;  // uniform wave id
  const int wRow = wv & 3;                                 // 4 row-waves * 32
  const int wCol = wv >> 2;                                // 2 col-waves * 64

  const int mBase = blockIdx.y * BM;
  const int nBase = blockIdx.x * BN;

  const unsigned smA0 = (unsigned)(unsigned long long)&smA[0][0];
  const unsigned smB0 = (unsigned)(unsigned long long)&smB[0][0];
  const unsigned sliceBytes = 16u * LDSS * 2u;    // one wave's 16-row slice
  const unsigned bufBytes   = (unsigned)(BM * LDSS * 2);

  // each wave DMAs its own 16-row slice of the A tile and of the B tile
  auto issue = [&](int buf, int k0) {
    tdm_load_tile_bf16(A + (size_t)(mBase + wv * 16) * K + k0,
                       smA0 + (unsigned)buf * bufBytes + (unsigned)wv * sliceBytes,
                       (unsigned)M, (unsigned)K);
    tdm_load_tile_bf16(W + (size_t)(nBase + wv * 16) * K + k0,
                       smB0 + (unsigned)buf * bufBytes + (unsigned)wv * sliceBytes,
                       (unsigned)N, (unsigned)K);
  };

  v8f acc[2][4] = {};

  issue(0, 0);
  __builtin_amdgcn_s_wait_tensorcnt(0);
  __syncthreads();

  const int KIT = K / BK;
  for (int it = 0; it < KIT; ++it) {
    const int buf = it & 1;
    if (it + 1 < KIT) issue(buf ^ 1, (it + 1) * BK);  // DMA next while computing

    // A fragments: lane m<16 needs K{0..7,16..23}; lane m+16 K{8..15,24..31}
    v16bf afrag[2];
#pragma unroll
    for (int mi = 0; mi < 2; ++mi) {
      const unsigned short* p =
          &smA[buf][(wRow * 32 + mi * 16 + l16) * LDSS + hk * 8];
      v8bf lo = *(const v8bf*)p;
      v8bf hi = *(const v8bf*)(p + 16);
      afrag[mi] = __builtin_shufflevector(lo, hi, 0, 1, 2, 3, 4, 5, 6, 7,
                                          8, 9, 10, 11, 12, 13, 14, 15);
    }
    // B fragments: lane n holds W row n (output feature), same K interleave
    v16bf bfrag[4];
#pragma unroll
    for (int ni = 0; ni < 4; ++ni) {
      const unsigned short* p =
          &smB[buf][(wCol * 64 + ni * 16 + l16) * LDSS + hk * 8];
      v8bf lo = *(const v8bf*)p;
      v8bf hi = *(const v8bf*)(p + 16);
      bfrag[ni] = __builtin_shufflevector(lo, hi, 0, 1, 2, 3, 4, 5, 6, 7,
                                          8, 9, 10, 11, 12, 13, 14, 15);
    }

#pragma unroll
    for (int mi = 0; mi < 2; ++mi)
#pragma unroll
      for (int ni = 0; ni < 4; ++ni)
        acc[mi][ni] = __builtin_amdgcn_wmma_f32_16x16x32_bf16(
            false, afrag[mi], false, bfrag[ni], (short)0, acc[mi][ni],
            false, false);

    if (it + 1 < KIT) {
      __builtin_amdgcn_s_wait_tensorcnt(0);  // own DMAs done
      __syncthreads();                       // everyone's DMAs done
    }
  }

  // writeout: C/D layout — lanes 0-15: M=r, N=lane; lanes 16-31: M=8+r
#pragma unroll
  for (int mi = 0; mi < 2; ++mi) {
#pragma unroll
    for (int ni = 0; ni < 4; ++ni) {
      int n_g = nBase + wCol * 64 + ni * 16 + l16;
      int m_g = mBase + wRow * 32 + mi * 16 + hk * 8;
      float bv = bias ? bias[n_g] : 0.0f;
#pragma unroll
      for (int r = 0; r < 8; ++r)
        C[(size_t)(m_g + r) * N + n_g] = acc[mi][ni][r] + bv;
    }
  }
}

// ------------------------------ GRU epilogue -------------------------------
__global__ void gru_fuse_kernel(const float* __restrict__ xproj,
                                const float* __restrict__ rzh,
                                const float* __restrict__ nh,
                                const float* __restrict__ h,
                                float* __restrict__ out) {
  int idx = blockIdx.x * blockDim.x + threadIdx.x;
  if (idx >= B_DIM * H_DIM) return;
  int i = idx >> 10;      // / H
  int j = idx & 1023;     // % H
  size_t x3 = (size_t)i * (3 * H_DIM);
  size_t h2 = (size_t)i * (2 * H_DIM);
  float r_x = xproj[x3 + j];
  float z_x = xproj[x3 + H_DIM + j];
  float n_x = xproj[x3 + 2 * H_DIM + j];
  float r_h = rzh[h2 + j];
  float z_h = rzh[h2 + H_DIM + j];
  float nhv = nh[idx];                       // already includes b_nh
  float r = 1.0f / (1.0f + __expf(-(r_x + r_h)));
  float z = 1.0f / (1.0f + __expf(-(z_x + z_h)));
  float n = tanhf(n_x + r * nhv);
  out[idx] = (1.0f - z) * n + z * h[idx];
}

// ------------------------------- launcher ----------------------------------
extern "C" void kernel_launch(void* const* d_in, const int* in_sizes, int n_in,
                              void* d_out, int out_size, void* d_ws, size_t ws_size,
                              hipStream_t stream) {
  (void)in_sizes; (void)n_in; (void)out_size; (void)ws_size;

  const float* x    = (const float*)d_in[0];
  const float* h    = (const float*)d_in[1];
  const float* Wih  = (const float*)d_in[2];
  const float* bih  = (const float*)d_in[3];
  const float* Wrzh = (const float*)d_in[4];
  const float* Wnh  = (const float*)d_in[5];
  const float* bnh  = (const float*)d_in[6];
  float* out = (float*)d_out;

  // workspace layout (bytes)
  char* ws = (char*)d_ws;
  constexpr size_t SZ_XBF   = (size_t)B_DIM * IN_DIM * 2;          // 8 MiB
  constexpr size_t SZ_HBF   = (size_t)B_DIM * H_DIM * 2;           // 8 MiB
  constexpr size_t SZ_WIHBF = (size_t)3 * H_DIM * IN_DIM * 2;      // 6 MiB
  constexpr size_t SZ_WRZBF = (size_t)2 * H_DIM * H_DIM * 2;       // 4 MiB
  constexpr size_t SZ_WNHBF = (size_t)H_DIM * H_DIM * 2;           // 2 MiB
  constexpr size_t SZ_XPROJ = (size_t)B_DIM * 3 * H_DIM * 4;       // 48 MiB
  constexpr size_t SZ_RZH   = (size_t)B_DIM * 2 * H_DIM * 4;       // 32 MiB

  size_t off = 0;
  unsigned short* x_bf   = (unsigned short*)(ws + off); off += SZ_XBF;
  unsigned short* h_bf   = (unsigned short*)(ws + off); off += SZ_HBF;
  unsigned short* Wih_bf = (unsigned short*)(ws + off); off += SZ_WIHBF;
  unsigned short* Wrz_bf = (unsigned short*)(ws + off); off += SZ_WRZBF;
  unsigned short* Wnh_bf = (unsigned short*)(ws + off); off += SZ_WNHBF;
  float* xproj = (float*)(ws + off); off += SZ_XPROJ;
  float* rzh   = (float*)(ws + off); off += SZ_RZH;
  float* nhbuf = (float*)(ws + off);

  auto conv = [&](const float* in, unsigned short* o, size_t n) {
    int n4 = (int)(n / 4);
    convert_f32_bf16_kernel<<<(n4 + 255) / 256, 256, 0, stream>>>(in, o, n4);
  };
  conv(x,    x_bf,   (size_t)B_DIM * IN_DIM);
  conv(h,    h_bf,   (size_t)B_DIM * H_DIM);
  conv(Wih,  Wih_bf, (size_t)3 * H_DIM * IN_DIM);
  conv(Wrzh, Wrz_bf, (size_t)2 * H_DIM * H_DIM);
  conv(Wnh,  Wnh_bf, (size_t)H_DIM * H_DIM);

  // GEMMs (grid: N/128 x M/128)
  gemm_bf16_wmma_kernel<<<dim3(3 * H_DIM / BN, B_DIM / BM), 256, 0, stream>>>(
      x_bf, Wih_bf, bih, xproj, B_DIM, 3 * H_DIM, IN_DIM);
  gemm_bf16_wmma_kernel<<<dim3(2 * H_DIM / BN, B_DIM / BM), 256, 0, stream>>>(
      h_bf, Wrz_bf, nullptr, rzh, B_DIM, 2 * H_DIM, H_DIM);
  gemm_bf16_wmma_kernel<<<dim3(H_DIM / BN, B_DIM / BM), 256, 0, stream>>>(
      h_bf, Wnh_bf, bnh, nhbuf, B_DIM, H_DIM, H_DIM);

  // fused GRU epilogue
  gru_fuse_kernel<<<(B_DIM * H_DIM) / 256, 256, 0, stream>>>(
      xproj, rzh, nhbuf, h, out);
}
